// MultiStageDiscriminator_33655363731806
// MI455X (gfx1250) — matrix operand order repaired and checked by reading, not compiled, not run on hardware
//
#include <hip/hip_runtime.h>
#include <math.h>

// ---------------------------------------------------------------------------
// Types for CDNA5 WMMA (wave32, 16x16x32 f16 -> f32)
// ---------------------------------------------------------------------------
typedef _Float16 half_t;
typedef __attribute__((ext_vector_type(2)))  _Float16 h2;
typedef __attribute__((ext_vector_type(8)))  _Float16 v8h;
typedef __attribute__((ext_vector_type(16))) _Float16 v16h;
typedef __attribute__((ext_vector_type(8)))  float    v8f;

__device__ __forceinline__ v8f wmma_f16(v16h a, v16h b, v8f c) {
    // D = A(16x32 f16) x B(32x16 f16) + C(16x16 f32)
    return __builtin_amdgcn_wmma_f32_16x16x32_f16(
        false, a, false, b, (short)0, c, false, false);
}

__device__ __forceinline__ v16h pack16(v8h lo, v8h hi) {
    v16h r;
#pragma unroll
    for (int j = 0; j < 8; ++j) { r[j] = lo[j]; r[j + 8] = hi[j]; }
    return r;
}

// Fragment layouts used here (A/B: same contiguous k order):
//  A (16x32, f16): lane L -> row M=L&15; halves j: K=akb+j;
//                  akb=(L<16)?0:16   -> two aligned 8-half runs.
//  B (32x16, f16): lane L -> col N=L&15; halves j: K=bkb+j; bkb=(L<16)?0:16.
//  C/D (16x16 f32): lane L -> col N=L&15; reg r -> row M=r+((L<16)?0:8).

#define EPS   1e-5f
#define SLOPE 0.2f
#define NB    16   // batch
#define NS    4    // stages

// ---------------------------------------------------------------------------
__global__ void cvt_f32_f16_kernel(const float* __restrict__ src,
                                   half_t* __restrict__ dst, int n) {
    int i = blockIdx.x * blockDim.x + threadIdx.x;
    if (i < n) dst[i] = (half_t)src[i];
}

__global__ void fill_zero_h_kernel(half_t* __restrict__ p, int n) {
    int i = blockIdx.x * blockDim.x + threadIdx.x;
    if (i < n) p[i] = (half_t)0.f;
}

// ---------------------------------------------------------------------------
// conv0: 1->64, k4 s2 p1, + bias + leaky.
// img (B,1,128,128) -> act0 f16 padded (B,64,66,66) interior [1..64]
// grid.x = B*64, block = 256
// ---------------------------------------------------------------------------
__global__ void conv0_kernel(const float* __restrict__ img,
                             const float* __restrict__ w0,   // (S,64,1,4,4)
                             const float* __restrict__ b0,   // (S,64)
                             half_t* __restrict__ act0,      // (B,64,66,66)
                             const int* __restrict__ stage_idx) {
    const int bc = blockIdx.x;
    const int b  = bc >> 6;
    const int c  = bc & 63;
    const int s  = stage_idx[b];
    __shared__ float wsm[16];
    __shared__ float bsm;
    if (threadIdx.x < 16) wsm[threadIdx.x] = w0[(s * 64 + c) * 16 + threadIdx.x];
    if (threadIdx.x == 0) bsm = b0[s * 64 + c];
    __syncthreads();
    const float* im = img + (size_t)b * 128 * 128;
    half_t* op = act0 + (size_t)bc * 66 * 66;
    for (int p = threadIdx.x; p < 64 * 64; p += blockDim.x) {
        const int oh = p >> 6, ow = p & 63;
        float sum = bsm;
#pragma unroll
        for (int kh = 0; kh < 4; ++kh) {
            const int ih = oh * 2 - 1 + kh;
            if ((unsigned)ih >= 128u) continue;
#pragma unroll
            for (int kw = 0; kw < 4; ++kw) {
                const int iw = ow * 2 - 1 + kw;
                if ((unsigned)iw < 128u)
                    sum += wsm[kh * 4 + kw] * im[ih * 128 + iw];
            }
        }
        sum = sum > 0.f ? sum : SLOPE * sum;
        op[(oh + 1) * 66 + (ow + 1)] = (half_t)sum;
    }
}

// ---------------------------------------------------------------------------
// Implicit-GEMM conv (k4 s2 p1) via WMMA on zero-padded input. No bounds checks.
// Out[b][m][n] = sum_k W[s][m][k] * patch(X)[k][n] + bias,  k = c*16+kh*4+kw
// Wave computes MT x NT grid of 16x16 tiles.  grid=(N/(16*NT), M/(16*MT), B), block=32
// ---------------------------------------------------------------------------
template <int CIN, int IHP, int IWP, int OH, int OW, int M, int MT, int NT>
__global__ void conv_gemm_wmma_kernel(const half_t* __restrict__ Wh,  // (S,M,K)
                                      const float*  __restrict__ bias,// (S,M)
                                      const half_t* __restrict__ Xp,  // (B,CIN,IHP,IWP)
                                      float* __restrict__ Out,        // (B,M,OH*OW)
                                      const int* __restrict__ stage_idx) {
    constexpr int K = CIN * 16;
    constexpr int N = OH * OW;
    const int lane = threadIdx.x;
    const int nblk = blockIdx.x;
    const int mblk = blockIdx.y;
    const int b    = blockIdx.z;
    const int s    = stage_idx[b];
    const half_t* W  = Wh + (size_t)s * M * K;
    const half_t* Xb = Xp + (size_t)b * CIN * IHP * IWP;

    const bool hi   = lane >= 16;
    const int  akb  = hi ? 16 : 0;
    const int  bkb  = hi ? 16 : 0;
    const int  lcol = lane & 15;

    int oh[NT], ow[NT];
#pragma unroll
    for (int t = 0; t < NT; ++t) {
        const int n = (nblk * NT + t) * 16 + lcol;
        oh[t] = n / OW;
        ow[t] = n % OW;
    }

    v8f acc[MT][NT];
#pragma unroll
    for (int i = 0; i < MT; ++i)
#pragma unroll
        for (int t = 0; t < NT; ++t) acc[i][t] = (v8f){};

    for (int kk = 0; kk < K; kk += 32) {
        // A fragments: two aligned 8-half vector loads each
        v16h a[MT];
#pragma unroll
        for (int i = 0; i < MT; ++i) {
            const half_t* wrow =
                W + (size_t)((mblk * MT + i) * 16 + lcol) * K + kk + akb;
            a[i] = pack16(*(const v8h*)(wrow), *(const v8h*)(wrow + 8));
        }
        // B fragments: one 4x4 patch of channel c, rows as 2x aligned h2 loads
        const int c = (kk + bkb) >> 4;
        const half_t* xc = Xb + (size_t)c * IHP * IWP;
        v16h bf[NT];
#pragma unroll
        for (int t = 0; t < NT; ++t) {
#pragma unroll
            for (int kh = 0; kh < 4; ++kh) {
                const h2* rp =
                    (const h2*)(xc + (size_t)(oh[t] * 2 + kh) * IWP + ow[t] * 2);
                const h2 p0 = rp[0], p1 = rp[1];
                bf[t][kh * 4 + 0] = p0[0];
                bf[t][kh * 4 + 1] = p0[1];
                bf[t][kh * 4 + 2] = p1[0];
                bf[t][kh * 4 + 3] = p1[1];
            }
        }
#pragma unroll
        for (int i = 0; i < MT; ++i)
#pragma unroll
            for (int t = 0; t < NT; ++t)
                acc[i][t] = wmma_f16(a[i], bf[t], acc[i][t]);
    }
    float* ob = Out + (size_t)b * M * N;
#pragma unroll
    for (int i = 0; i < MT; ++i) {
#pragma unroll
        for (int t = 0; t < NT; ++t) {
            const int n = (nblk * NT + t) * 16 + lcol;
#pragma unroll
            for (int r = 0; r < 8; ++r) {
                const int m = (mblk * MT + i) * 16 + r + (hi ? 8 : 0);
                ob[(size_t)m * N + n] = acc[i][t][r] + bias[s * M + m];
            }
        }
    }
}

// ---------------------------------------------------------------------------
// 1x1-conv GEMM via WMMA, f16 output, MTxNT blocking.
// X is TOKEN-MAJOR: XT (B,N,K)  -> B-fragment = two aligned v8h loads.
// Output written with generic strides: Out[b*outB + m*rs + n*cs]
//   (rs=N, cs=1  -> channel-major;  rs=1, cs=32 -> token-major q/k with pad)
// grid = (N/(16*NT), M/(16*MT), B), block = 32
// ---------------------------------------------------------------------------
template <int MT, int NT>
__global__ void gemm1x1_wmma_kernel(const half_t* __restrict__ Wh,   // (S,M,K)
                                    const float*  __restrict__ bias, // (S,M)
                                    const half_t* __restrict__ XT,   // (B,N,K)
                                    half_t* __restrict__ Out,
                                    const int* __restrict__ stage_idx,
                                    int M, int K, int N,
                                    int outB, int rs, int cs) {
    const int lane = threadIdx.x;
    const int nblk = blockIdx.x;
    const int mblk = blockIdx.y;
    const int b    = blockIdx.z;
    const int s    = stage_idx[b];
    const half_t* W  = Wh + (size_t)s * M * K;
    const half_t* Xb = XT + (size_t)b * N * K;

    const bool hi   = lane >= 16;
    const int  akb  = hi ? 16 : 0;
    const int  bkb  = hi ? 16 : 0;
    const int  lcol = lane & 15;

    v8f acc[MT][NT];
#pragma unroll
    for (int i = 0; i < MT; ++i)
#pragma unroll
        for (int t = 0; t < NT; ++t) acc[i][t] = (v8f){};

    for (int kk = 0; kk < K; kk += 32) {
        v16h a[MT];
#pragma unroll
        for (int i = 0; i < MT; ++i) {
            const half_t* wrow =
                W + (size_t)((mblk * MT + i) * 16 + lcol) * K + kk + akb;
            a[i] = pack16(*(const v8h*)(wrow), *(const v8h*)(wrow + 8));
        }
        v16h bf[NT];
#pragma unroll
        for (int t = 0; t < NT; ++t) {
            const int n = (nblk * NT + t) * 16 + lcol;
            const half_t* xp = Xb + (size_t)n * K + kk + bkb;  // 16 contiguous ch
            bf[t] = pack16(*(const v8h*)(xp), *(const v8h*)(xp + 8));
        }
#pragma unroll
        for (int i = 0; i < MT; ++i)
#pragma unroll
            for (int t = 0; t < NT; ++t)
                acc[i][t] = wmma_f16(a[i], bf[t], acc[i][t]);
    }
    half_t* ob = Out + (size_t)b * outB;
#pragma unroll
    for (int i = 0; i < MT; ++i) {
#pragma unroll
        for (int t = 0; t < NT; ++t) {
            const int n = (nblk * NT + t) * 16 + lcol;
#pragma unroll
            for (int r = 0; r < 8; ++r) {
                const int m = (mblk * MT + i) * 16 + r + (hi ? 8 : 0);
                ob[(size_t)m * rs + (size_t)n * cs] =
                    (half_t)(acc[i][t][r] + bias[s * M + m]);
            }
        }
    }
}

// ---------------------------------------------------------------------------
// Instance norm (biased var, eps=1e-5) + leaky relu, in place;
// optional TRANSPOSED f16 shadow: xT16[b][i][c], c-dim length C.
// grid.x = B*C, block = 256
// ---------------------------------------------------------------------------
__global__ void inorm_leaky_kernel(float* __restrict__ x,
                                   half_t* __restrict__ xT16, int N, int C) {
    const int bc = blockIdx.x;
    const int b  = bc / C;
    const int c  = bc % C;
    float* xp = x + (size_t)bc * N;
    __shared__ float s1buf[256], s2buf[256];
    float s1 = 0.f, s2 = 0.f;
    for (int i = threadIdx.x; i < N; i += blockDim.x) {
        const float v = xp[i];
        s1 += v;
        s2 += v * v;
    }
    s1buf[threadIdx.x] = s1;
    s2buf[threadIdx.x] = s2;
    __syncthreads();
    for (int st = 128; st > 0; st >>= 1) {
        if (threadIdx.x < st) {
            s1buf[threadIdx.x] += s1buf[threadIdx.x + st];
            s2buf[threadIdx.x] += s2buf[threadIdx.x + st];
        }
        __syncthreads();
    }
    const float mean = s1buf[0] / (float)N;
    const float var  = s2buf[0] / (float)N - mean * mean;
    const float rstd = rsqrtf(var + EPS);
    for (int i = threadIdx.x; i < N; i += blockDim.x) {
        float v = (xp[i] - mean) * rstd;
        v = v > 0.f ? v : SLOPE * v;
        xp[i] = v;
        if (xT16) xT16[((size_t)b * N + i) * C + c] = (half_t)v;
    }
}

// ---------------------------------------------------------------------------
// Self-attention; attention matrix lives in LDS only.
// q/k stored TOKEN-MAJOR (B,1024,32), channels 16..31 zero -> all fragment
// loads are aligned v8h pairs.
//   e[n][m] = sum_c q[n][c] k[m][c]
//   P       = softmax_m(e)
//   y[c][h+1][w+1] = gamma * sum_m P[n][m] v[c][m] + x[c][n]   (padded store)
// grid = (64 query blocks, 1, B), block = 128 (4 waves)
// ---------------------------------------------------------------------------
#define ATTN_SMEM_BYTES (65536 + 32768 + 64 + 64 + 512)

__global__ void attn_kernel(const half_t* __restrict__ qT,   // (B,1024,32)
                            const half_t* __restrict__ kT,   // (B,1024,32)
                            const half_t* __restrict__ vh,   // (B,128,1024)
                            const float*  __restrict__ xres, // (B,128,1024) act1 f32
                            const float*  __restrict__ gamma,// (S,)
                            half_t* __restrict__ yh,         // (B,128,34,34) padded
                            const int* __restrict__ stage_idx) {
    extern __shared__ char smem[];
    float*  e      = (float*)smem;                     // 16 x 1024
    half_t* pl     = (half_t*)(smem + 65536);          // 16 x 1024
    float*  rowmax = (float*)(smem + 65536 + 32768);   // 16
    float*  rowsum = rowmax + 16;                      // 16
    float*  part   = rowsum + 16;                      // 16 x 8

    const int nb   = blockIdx.x;
    const int b    = blockIdx.z;
    const int tid  = threadIdx.x;
    const int wave = tid >> 5;
    const int lane = tid & 31;
    const int s    = stage_idx[b];

    const half_t* q = qT + (size_t)b * 1024 * 32;
    const half_t* k = kT + (size_t)b * 1024 * 32;
    const half_t* v = vh + (size_t)b * 128 * 1024;
    const float*  x = xres + (size_t)b * 128 * 1024;

    const int  lcol = lane & 15;
    const bool hi   = lane >= 16;
    const int  akb  = hi ? 16 : 0;
    const int  bkb  = hi ? 16 : 0;

    // q A-fragment (rows = queries, K = 32 channels; upper 16 are stored zeros)
    v16h aq;
    {
        const half_t* qp = q + (size_t)(nb * 16 + lcol) * 32 + akb;
        aq = pack16(*(const v8h*)(qp), *(const v8h*)(qp + 8));
    }

    // ---- energies into LDS ----
    for (int t = wave; t < 64; t += 4) {
        const half_t* kp = k + (size_t)(t * 16 + lcol) * 32 + bkb;
        const v16h bk = pack16(*(const v8h*)(kp), *(const v8h*)(kp + 8));
        v8f d = {};
        d = wmma_f16(aq, bk, d);
#pragma unroll
        for (int r = 0; r < 8; ++r) {
            const int qr = r + (hi ? 8 : 0);
            e[qr * 1024 + t * 16 + lcol] = d[r];
        }
    }
    __syncthreads();

    // ---- exact softmax over m, 8 threads per row ----
    {
        const int qr = tid >> 3, sub = tid & 7;
        float mx = -3.0e38f;
        for (int m = sub; m < 1024; m += 8) mx = fmaxf(mx, e[qr * 1024 + m]);
        part[qr * 8 + sub] = mx;
        __syncthreads();
        if (sub == 0) {
            float m2 = part[qr * 8];
#pragma unroll
            for (int j = 1; j < 8; ++j) m2 = fmaxf(m2, part[qr * 8 + j]);
            rowmax[qr] = m2;
        }
        __syncthreads();
        const float rm = rowmax[qr];
        float sum = 0.f;
        for (int m = sub; m < 1024; m += 8) {
            const float ev = __expf(e[qr * 1024 + m] - rm);
            e[qr * 1024 + m] = ev;
            sum += ev;
        }
        part[qr * 8 + sub] = sum;
        __syncthreads();
        if (sub == 0) {
            float t2 = part[qr * 8];
#pragma unroll
            for (int j = 1; j < 8; ++j) t2 += part[qr * 8 + j];
            rowsum[qr] = t2;
        }
        __syncthreads();
        const float inv = 1.f / rowsum[qr];
        for (int m = sub; m < 1024; m += 8)
            pl[qr * 1024 + m] = (half_t)(e[qr * 1024 + m] * inv);
    }
    __syncthreads();

    // ---- P x V^T (out[q][c]); A fragment shared across two c-tiles ----
    const float g = gamma[s];
    v8f acc[2] = {(v8f){}, (v8f){}};
    for (int mc = 0; mc < 1024; mc += 32) {
        const half_t* pp = pl + (lane & 15) * 1024 + mc + akb;
        const v16h ap = pack16(*(const v8h*)(pp), *(const v8h*)(pp + 8));
#pragma unroll
        for (int u = 0; u < 2; ++u) {
            const int c = (wave + 4 * u) * 16 + lcol;
            const half_t* vp = v + (size_t)c * 1024 + mc + bkb;
            const v16h bv = pack16(*(const v8h*)(vp), *(const v8h*)(vp + 8));
            acc[u] = wmma_f16(ap, bv, acc[u]);
        }
    }
#pragma unroll
    for (int u = 0; u < 2; ++u) {
        const int c = (wave + 4 * u) * 16 + lcol;
#pragma unroll
        for (int r = 0; r < 8; ++r) {
            const int qrow = r + (hi ? 8 : 0);
            const int n = nb * 16 + qrow;
            const int h = n >> 5, w = n & 31;
            yh[((size_t)b * 128 + c) * 34 * 34 + (h + 1) * 34 + (w + 1)] =
                (half_t)(g * acc[u][r] + x[(size_t)c * 1024 + n]);
        }
    }
}

// ---------------------------------------------------------------------------
// Head conv: 256->1, k4 s1 p1 on 16x16 -> 15x15.  grid.x = B*225, block = 128
// ---------------------------------------------------------------------------
__global__ void head_kernel(const float* __restrict__ act2, // (B,256,16,16)
                            const float* __restrict__ wh,   // (S,1,256,4,4)
                            const float* __restrict__ bh,   // (S,1)
                            float* __restrict__ out,        // (B,1,15,15)
                            const int* __restrict__ stage_idx) {
    const int b    = blockIdx.x / 225;
    const int pidx = blockIdx.x % 225;
    const int oh = pidx / 15, ow = pidx % 15;
    const int s = stage_idx[b];
    const float* xb = act2 + (size_t)b * 256 * 256;
    const float* w  = wh + (size_t)s * 256 * 16;
    float sum = 0.f;
    for (int i = threadIdx.x; i < 4096; i += blockDim.x) {
        const int c  = i >> 4;
        const int kh = (i >> 2) & 3;
        const int kw = i & 3;
        const int ih = oh - 1 + kh;
        const int iw = ow - 1 + kw;
        if ((unsigned)ih < 16u && (unsigned)iw < 16u)
            sum += w[c * 16 + kh * 4 + kw] * xb[c * 256 + ih * 16 + iw];
    }
    __shared__ float red[128];
    red[threadIdx.x] = sum;
    __syncthreads();
    for (int st = 64; st > 0; st >>= 1) {
        if (threadIdx.x < st) red[threadIdx.x] += red[threadIdx.x + st];
        __syncthreads();
    }
    if (threadIdx.x == 0) out[(size_t)b * 225 + pidx] = red[0] + bh[s];
}

// ---------------------------------------------------------------------------
// launch
// ---------------------------------------------------------------------------
static inline size_t alignUp(size_t x) { return (x + 255) & ~(size_t)255; }

extern "C" void kernel_launch(void* const* d_in, const int* in_sizes, int n_in,
                              void* d_out, int out_size, void* d_ws, size_t ws_size,
                              hipStream_t stream) {
    (void)in_sizes; (void)n_in; (void)out_size; (void)ws_size;

    const float* img  = (const float*)d_in[0];
    const int*   sidx = (const int*)d_in[1];
    const float* w0   = (const float*)d_in[2];
    const float* b0   = (const float*)d_in[3];
    const float* w1   = (const float*)d_in[4];
    const float* b1   = (const float*)d_in[5];
    const float* w2   = (const float*)d_in[6];
    const float* b2   = (const float*)d_in[7];
    const float* wq   = (const float*)d_in[8];
    const float* bq   = (const float*)d_in[9];
    const float* wk   = (const float*)d_in[10];
    const float* bk   = (const float*)d_in[11];
    const float* wv   = (const float*)d_in[12];
    const float* bv   = (const float*)d_in[13];
    const float* gm   = (const float*)d_in[14];
    const float* whd  = (const float*)d_in[15];
    const float* bhd  = (const float*)d_in[16];
    float* out = (float*)d_out;

    // ---- workspace carve-out ----
    char* ws = (char*)d_ws;
    size_t off = 0;
    auto take = [&](size_t bytes) { char* p = ws + off; off = alignUp(off + bytes); return p; };

    const size_t n_w1 = (size_t)NS * 128 * 64 * 16;
    const size_t n_w2 = (size_t)NS * 256 * 128 * 16;
    const size_t n_wq = (size_t)NS * 16 * 128;
    const size_t n_wv = (size_t)NS * 128 * 128;
    const int n_act0 = NB * 64 * 66 * 66;   // padded
    const int n_ypad = NB * 128 * 34 * 34;  // padded
    const int n_qk   = NB * 1024 * 32;      // token-major, channel-padded

    half_t* w1h   = (half_t*)take(n_w1 * 2);
    half_t* w2h   = (half_t*)take(n_w2 * 2);
    half_t* wqh   = (half_t*)take(n_wq * 2);
    half_t* wkh   = (half_t*)take(n_wq * 2);
    half_t* wvh   = (half_t*)take(n_wv * 2);
    half_t* act0p = (half_t*)take((size_t)n_act0 * 2);
    float*  c1    = (float*)take((size_t)NB * 128 * 1024 * 4);
    half_t* a1T   = (half_t*)take((size_t)NB * 1024 * 128 * 2);  // token-major
    half_t* qT    = (half_t*)take((size_t)n_qk * 2);
    half_t* kT    = (half_t*)take((size_t)n_qk * 2);
    half_t* vh    = (half_t*)take((size_t)NB * 128 * 1024 * 2);
    half_t* ypad  = (half_t*)take((size_t)n_ypad * 2);
    float*  c2    = (float*)take((size_t)NB * 256 * 256 * 4);

    // ---- 1. weight conversion + zero-fill padded buffers ----
    cvt_f32_f16_kernel<<<(int)((n_w1 + 255) / 256), 256, 0, stream>>>(w1, w1h, (int)n_w1);
    cvt_f32_f16_kernel<<<(int)((n_w2 + 255) / 256), 256, 0, stream>>>(w2, w2h, (int)n_w2);
    cvt_f32_f16_kernel<<<(int)((n_wq + 255) / 256), 256, 0, stream>>>(wq, wqh, (int)n_wq);
    cvt_f32_f16_kernel<<<(int)((n_wq + 255) / 256), 256, 0, stream>>>(wk, wkh, (int)n_wq);
    cvt_f32_f16_kernel<<<(int)((n_wv + 255) / 256), 256, 0, stream>>>(wv, wvh, (int)n_wv);
    fill_zero_h_kernel<<<(n_act0 + 255) / 256, 256, 0, stream>>>(act0p, n_act0);
    fill_zero_h_kernel<<<(n_ypad + 255) / 256, 256, 0, stream>>>(ypad, n_ypad);
    fill_zero_h_kernel<<<(n_qk + 255) / 256, 256, 0, stream>>>(qT, n_qk);
    fill_zero_h_kernel<<<(n_qk + 255) / 256, 256, 0, stream>>>(kT, n_qk);

    // ---- 2. conv0 + leaky (writes interior of padded act0) ----
    conv0_kernel<<<NB * 64, 256, 0, stream>>>(img, w0, b0, act0p, sidx);

    // ---- 3. conv1 (implicit GEMM, WMMA): M=128, K=1024, N=1024 ----
    conv_gemm_wmma_kernel<64, 66, 66, 32, 32, 128, 2, 4>
        <<<dim3(16, 4, NB), 32, 0, stream>>>(w1h, b1, act0p, c1, sidx);

    // ---- 4. instance norm + leaky (in place) + token-major f16 shadow ----
    inorm_leaky_kernel<<<NB * 128, 256, 0, stream>>>(c1, a1T, 1024, 128);

    // ---- 5. q/k/v projections (WMMA); q/k stored token-major with 32-ch pad ----
    gemm1x1_wmma_kernel<1, 4><<<dim3(16, 1, NB), 32, 0, stream>>>(
        wqh, bq, a1T, qT, sidx, 16, 128, 1024, 1024 * 32, /*rs=*/1, /*cs=*/32);
    gemm1x1_wmma_kernel<1, 4><<<dim3(16, 1, NB), 32, 0, stream>>>(
        wkh, bk, a1T, kT, sidx, 16, 128, 1024, 1024 * 32, /*rs=*/1, /*cs=*/32);
    gemm1x1_wmma_kernel<2, 4><<<dim3(16, 4, NB), 32, 0, stream>>>(
        wvh, bv, a1T, vh, sidx, 128, 128, 1024, 128 * 1024, /*rs=*/1024, /*cs=*/1);

    // ---- 6. self-attention (LDS-resident attention matrix, WMMA both matmuls) ----
    attn_kernel<<<dim3(64, 1, NB), 128, ATTN_SMEM_BYTES, stream>>>(
        qT, kT, vh, c1, gm, ypad, sidx);

    // ---- 7. conv2 (implicit GEMM, WMMA): M=256, K=2048, N=256 ----
    conv_gemm_wmma_kernel<128, 34, 34, 16, 16, 256, 2, 4>
        <<<dim3(4, 8, NB), 32, 0, stream>>>(w2h, b2, ypad, c2, sidx);

    // ---- 8. instance norm + leaky (in place) ----
    inorm_leaky_kernel<<<NB * 256, 256, 0, stream>>>(c2, (half_t*)nullptr, 256, 256);

    // ---- 9. head conv -> output (B,1,15,15) f32 ----
    head_kernel<<<NB * 225, 128, 0, stream>>>(c2, whd, bhd, out, sidx);
}
